// InterpretableMultiHeadAttention_45904610459999
// MI455X (gfx1250) — matrix operand order repaired and checked by reading, not compiled, not run on hardware
//
#include <hip/hip_runtime.h>
#include <hip/hip_bf16.h>

typedef __bf16 bf16;
typedef __attribute__((ext_vector_type(16))) __bf16 v16bf;
typedef __attribute__((ext_vector_type(8)))  float  v8f;
typedef unsigned int v4u __attribute__((ext_vector_type(4)));
typedef int          v8i __attribute__((ext_vector_type(8)));
typedef int          v4i __attribute__((ext_vector_type(4)));

constexpr int kB  = 2;
constexpr int kS  = 2048;
constexpr int kDM = 1024;
constexpr int kDK = 64;

// K-tile LDS layout produced by the TDM pad engine:
// 16 rows of 1024 bf16, with 16B pad inserted after every 1024B (256 dwords)
// => row stride 2080B = 1040 bf16, and columns >= 512 shift by +8 elements.
constexpr int kKRow = 1040;
__device__ __forceinline__ int k_coladj(int col) { return col + ((col >> 9) << 3); }

// Per-lane K index for a 16x32 bf16 A fragment element e (contiguous: k = 16*hlf + e).
__device__ __forceinline__ int a_kidx(int e, int hlf) {
  return hlf * 16 + e;
}

// ---------------------------------------------------------------------------
// Generic GEMM:  C[M,N] = A_f32[M,Kd] @ W_f32[Kd,N] + bias,  C in bf16 or f32.
// Block tile 128x64, 8 waves as 4(M)x2(N), each wave 2x2 WMMA 16x16 tiles.
// ---------------------------------------------------------------------------
template <bool OUT_BF16>
__global__ __launch_bounds__(256) void gemm_bf16_wmma(
    const float* __restrict__ A, const float* __restrict__ W,
    const float* __restrict__ bias, void* __restrict__ Cout,
    int M, int N, int Kd) {
  __shared__ bf16 At[128][40];  // [m][k], padded
  __shared__ bf16 Bt[64][40];   // [n][k], padded (W transposed on stage-in)

  const int t    = threadIdx.x;
  const int lane = t & 31;
  const int wave = t >> 5;
  const int hlf  = lane >> 4;
  const int ln   = lane & 15;
  const int wm   = wave >> 1;   // 0..3
  const int wn   = wave & 1;    // 0..1
  const int m0   = blockIdx.y * 128;
  const int n0   = blockIdx.x * 64;

  v8f acc[2][2] = {};

  for (int k0 = 0; k0 < Kd; k0 += 32) {
    __syncthreads();
    {  // stage A tile: 128 rows x 32 cols, float4 loads -> bf16 LDS
      int col = (t & 7) * 4;
      int r0  = t >> 3;
#pragma unroll
      for (int rr = 0; rr < 4; ++rr) {
        int row = r0 + rr * 32;
        const float4 x =
            *(const float4*)&A[(size_t)(m0 + row) * Kd + k0 + col];
        At[row][col + 0] = (bf16)x.x;
        At[row][col + 1] = (bf16)x.y;
        At[row][col + 2] = (bf16)x.z;
        At[row][col + 3] = (bf16)x.w;
      }
    }
    {  // stage B tile transposed: Bt[n][k] = W[k0+k][n0+n]
      int n  = t & 63;
      int kb = t >> 6;  // 0..3
#pragma unroll
      for (int kk = 0; kk < 8; ++kk) {
        int k = kb + kk * 4;
        Bt[n][k] = (bf16)W[(size_t)(k0 + k) * N + n0 + n];
      }
    }
    __syncthreads();

    v16bf af[2], bfr[2];
#pragma unroll
    for (int mt = 0; mt < 2; ++mt) {
      int mrow = wm * 32 + mt * 16 + ln;
#pragma unroll
      for (int e = 0; e < 16; ++e) af[mt][e] = At[mrow][a_kidx(e, hlf)];
    }
#pragma unroll
    for (int nt = 0; nt < 2; ++nt) {
      int ncol = wn * 32 + nt * 16 + ln;
#pragma unroll
      for (int e = 0; e < 16; ++e) bfr[nt][e] = Bt[ncol][hlf * 16 + e];
    }
#pragma unroll
    for (int mt = 0; mt < 2; ++mt)
#pragma unroll
      for (int nt = 0; nt < 2; ++nt)
        acc[mt][nt] = __builtin_amdgcn_wmma_f32_16x16x32_bf16(
            false, af[mt], false, bfr[nt], (short)0, acc[mt][nt], false, false);
  }

#pragma unroll
  for (int mt = 0; mt < 2; ++mt)
#pragma unroll
    for (int nt = 0; nt < 2; ++nt)
#pragma unroll
      for (int r = 0; r < 8; ++r) {
        int row = m0 + wm * 32 + mt * 16 + r + hlf * 8;
        int col = n0 + wn * 32 + nt * 16 + ln;
        float v = acc[mt][nt][r] + bias[col];
        if (OUT_BF16)
          ((bf16*)Cout)[(size_t)row * N + col] = (bf16)v;
        else
          ((float*)Cout)[(size_t)row * N + col] = v;
      }
}

// ---------------------------------------------------------------------------
// K-tile stage: one TDM descriptor (wave 0) or manual uint4 copies (fallback).
// Loads 16 key rows x 2048B from Kb into kbuf with the padded layout above.
// ---------------------------------------------------------------------------
__device__ __forceinline__ void stage_ktile(bf16* kbuf, const bf16* Kb,
                                            size_t row0, int t, int wave) {
#if __has_builtin(__builtin_amdgcn_tensor_load_to_lds)
  if (wave == 0) {  // wave-uniform; tensor ops ignore EXEC
    unsigned long long ga =
        (unsigned long long)(const void*)(Kb + row0 * kDM);
    unsigned int la = (unsigned int)(unsigned long long)(void*)kbuf;  // LDS off
    v4u g0;
    g0.x = 1u;                                   // count=1 (valid user D#)
    g0.y = la;                                   // lds_addr
    g0.z = (unsigned int)ga;                     // global_addr[31:0]
    g0.w = ((unsigned int)(ga >> 32) & 0x01FFFFFFu) | (2u << 30);  // type=2
    v8i g1;
    g1[0] = (2 << 16)      // data_size = 4B
          | (1 << 20)      // pad_enable
          | (7 << 22)      // pad_interval: every 256 dwords (1024B)
          | (3 << 25);     // pad_amount: 4 dwords (16B)
    g1[1] = 512 << 16;     // tensor_dim0 = 512 dwords (low 16)
    g1[2] = 4096 << 16;    // tensor_dim0 hi=0 | tensor_dim1 = 4096 rows
    g1[3] = 512 << 16;     // tensor_dim1 hi=0 | tile_dim0 = 512 dwords
    g1[4] = 16;            // tile_dim1 = 16 rows, tile_dim2 = 0
    g1[5] = 512;           // tensor_dim0_stride = 512 dwords
    g1[6] = 0;
    g1[7] = 0;
    v4i z4 = {0, 0, 0, 0};
#if __clang_major__ >= 23
    v8i z8 = {0, 0, 0, 0, 0, 0, 0, 0};
    __builtin_amdgcn_tensor_load_to_lds(g0, g1, z4, z4, z8, 0);
#else
    __builtin_amdgcn_tensor_load_to_lds(g0, g1, z4, z4, 0);
#endif
    __builtin_amdgcn_s_wait_tensorcnt(0);
  }
#else
  int key = t >> 4, c0 = (t & 15) * 64;
  const uint4* src = (const uint4*)&Kb[(row0 + key) * kDM + c0];
  uint4* dst = (uint4*)&kbuf[key * kKRow + k_coladj(c0)];
#pragma unroll
  for (int j = 0; j < 8; ++j) dst[j] = src[j];
#endif
}

// ---------------------------------------------------------------------------
// Fused attention. One block = 16 query rows of one batch.
// 8 waves; wave w owns heads 2w, 2w+1. Two passes over keys:
//   pass1: online softmax stats (row max / sumexp) per head, kept in regs.
//   pass2: probs -> head-average (LDS ds_add_f32) -> write avg_attn,
//          every 32 keys: WMMA  avg(bf16) @ V  into register accumulators.
// ---------------------------------------------------------------------------
__global__ __launch_bounds__(256) void attn_fused(
    const bf16* __restrict__ Qb, const bf16* __restrict__ Kb,
    const bf16* __restrict__ Vb, float* __restrict__ avg_out,
    float* __restrict__ attn_out) {
  __shared__ bf16  kbuf[16 * kKRow];  // 16 keys x all-head d_k, TDM-padded
  __shared__ bf16  vtile[64][40];     // [d_k][32-key window], padded
  __shared__ float avgbuf[16][34];    // head-averaged probs, [row][32 keys]

  const int t    = threadIdx.x;
  const int lane = t & 31;
  const int wave = t >> 5;
  const int hlf  = lane >> 4;
  const int ln   = lane & 15;
  const int b    = blockIdx.x >> 7;         // 128 q-blocks per batch
  const int q0   = (blockIdx.x & 127) * 16;
  const int h0   = wave * 2;
  const float scale = 0.125f;               // 1/sqrt(d_k)

  // Q fragments for this wave's two heads (K=64 -> 2 k-steps), held in regs.
  v16bf qf[2][2];
#pragma unroll
  for (int hl = 0; hl < 2; ++hl)
#pragma unroll
    for (int ks = 0; ks < 2; ++ks)
#pragma unroll
      for (int e = 0; e < 16; ++e) {
        int col = (h0 + hl) * kDK + ks * 32 + a_kidx(e, hlf);
        qf[hl][ks][e] = Qb[(size_t)(b * kS + q0 + ln) * kDM + col];
      }

  float mrun[2][8], srun[2][8];
#pragma unroll
  for (int hl = 0; hl < 2; ++hl)
#pragma unroll
    for (int r = 0; r < 8; ++r) { mrun[hl][r] = -1e30f; srun[hl][r] = 0.f; }

  // ---------------- pass 1: streaming row max & sumexp ----------------
  for (int it = 0; it < kS / 16; ++it) {
    __syncthreads();
    stage_ktile(kbuf, Kb, (size_t)(b * kS + it * 16), t, wave);
    if (it + 1 < kS / 16)  // prefetch next K tile (global_prefetch_b8)
      __builtin_prefetch(
          &Kb[(size_t)(b * kS + (it + 1) * 16 + (t >> 4)) * kDM +
              (t & 15) * 64], 0, 1);
    __syncthreads();
#pragma unroll
    for (int hl = 0; hl < 2; ++hl) {
      v8f c = {};
#pragma unroll
      for (int ks = 0; ks < 2; ++ks) {
        int bcol = (h0 + hl) * kDK + ks * 32;
        int cb = k_coladj(bcol) + hlf * 16;
        v16bf kf;
#pragma unroll
        for (int e = 0; e < 16; ++e) kf[e] = kbuf[ln * kKRow + cb + e];
        c = __builtin_amdgcn_wmma_f32_16x16x32_bf16(false, qf[hl][ks], false,
                                                    kf, (short)0, c, false,
                                                    false);
      }
#pragma unroll
      for (int r = 0; r < 8; ++r) {
        float v = c[r] * scale;
        float tmax = v;
#pragma unroll
        for (int off = 1; off < 16; off <<= 1)
          tmax = fmaxf(tmax, __shfl_xor(tmax, off, 32));
        float nm = fmaxf(mrun[hl][r], tmax);
        float ev = __expf(v - nm);
#pragma unroll
        for (int off = 1; off < 16; off <<= 1) ev += __shfl_xor(ev, off, 32);
        srun[hl][r] = srun[hl][r] * __expf(mrun[hl][r] - nm) + ev;
        mrun[hl][r] = nm;
      }
    }
  }
  float inv[2][8];
#pragma unroll
  for (int hl = 0; hl < 2; ++hl)
#pragma unroll
    for (int r = 0; r < 8; ++r) inv[hl][r] = 1.f / srun[hl][r];

  // ---------------- pass 2: probs -> avg_attn -> avg@V ----------------
  v8f oacc = {};  // waves 0..3: 16 rows x 16 d_k columns of attn_out
  for (int it = 0; it < kS / 16; ++it) {
    const int par = it & 1;
    __syncthreads();
    stage_ktile(kbuf, Kb, (size_t)(b * kS + it * 16), t, wave);
    {  // stage V tile (16 keys x 64 d_k) transposed into current window half
      int key = t >> 4, d0 = (t & 15) * 4;
#pragma unroll
      for (int j = 0; j < 4; ++j)
        vtile[d0 + j][par * 16 + key] =
            Vb[(size_t)(b * kS + it * 16 + key) * kDK + d0 + j];
    }
    avgbuf[t >> 4][par * 16 + (t & 15)] = 0.f;  // zero this half
    __syncthreads();

    float psum[8] = {0.f, 0.f, 0.f, 0.f, 0.f, 0.f, 0.f, 0.f};
#pragma unroll
    for (int hl = 0; hl < 2; ++hl) {
      v8f c = {};
#pragma unroll
      for (int ks = 0; ks < 2; ++ks) {
        int bcol = (h0 + hl) * kDK + ks * 32;
        int cb = k_coladj(bcol) + hlf * 16;
        v16bf kf;
#pragma unroll
        for (int e = 0; e < 16; ++e) kf[e] = kbuf[ln * kKRow + cb + e];
        c = __builtin_amdgcn_wmma_f32_16x16x32_bf16(false, qf[hl][ks], false,
                                                    kf, (short)0, c, false,
                                                    false);
      }
#pragma unroll
      for (int r = 0; r < 8; ++r)
        psum[r] += __expf(c[r] * scale - mrun[hl][r]) * inv[hl][r];
    }
#pragma unroll
    for (int r = 0; r < 8; ++r)
      atomicAdd(&avgbuf[r + hlf * 8][par * 16 + ln], psum[r] * 0.0625f);
    __syncthreads();

    {  // write head-averaged attention tile (exactly once)
      int rr = t >> 4, nn = t & 15;
      avg_out[(size_t)(b * kS + q0 + rr) * kS + it * 16 + nn] =
          avgbuf[rr][par * 16 + nn];
    }
    if (par == 1 && wave < 4) {  // wave-uniform: EXEC stays all-ones
      v16bf aaf, vf;
#pragma unroll
      for (int e = 0; e < 16; ++e) aaf[e] = (bf16)avgbuf[ln][a_kidx(e, hlf)];
#pragma unroll
      for (int e = 0; e < 16; ++e) vf[e] = vtile[wave * 16 + ln][hlf * 16 + e];
      oacc = __builtin_amdgcn_wmma_f32_16x16x32_bf16(false, aaf, false, vf,
                                                     (short)0, oacc, false,
                                                     false);
    }
  }
  if (wave < 4) {
#pragma unroll
    for (int r = 0; r < 8; ++r)
      attn_out[(size_t)(b * kS + q0 + r + hlf * 8) * kDK + wave * 16 + ln] =
          oacc[r];
  }
}

// ---------------------------------------------------------------------------
extern "C" void kernel_launch(void* const* d_in, const int* in_sizes, int n_in,
                              void* d_out, int out_size, void* d_ws,
                              size_t ws_size, hipStream_t stream) {
  (void)in_sizes; (void)n_in; (void)out_size; (void)ws_size;
  const float* query = (const float*)d_in[0];
  const float* key_  = (const float*)d_in[1];
  const float* value = (const float*)d_in[2];
  const float* Wq = (const float*)d_in[3];
  const float* bq = (const float*)d_in[4];
  const float* Wk = (const float*)d_in[5];
  const float* bk = (const float*)d_in[6];
  const float* Wv = (const float*)d_in[7];
  const float* bv = (const float*)d_in[8];
  const float* Wo = (const float*)d_in[9];
  const float* bo = (const float*)d_in[10];

  float* out = (float*)d_out;                        // [B,S,DM]
  float* avg = out + (size_t)kB * kS * kDM;          // [B,S,S]

  char* ws = (char*)d_ws;
  bf16* Qb = (bf16*)ws; ws += (size_t)kB * kS * kDM * sizeof(bf16);
  bf16* Kb = (bf16*)ws; ws += (size_t)kB * kS * kDM * sizeof(bf16);
  bf16* Vb = (bf16*)ws; ws += (size_t)kB * kS * kDK * sizeof(bf16);
  float* attn_out = (float*)ws;                      // [B*S, DK] f32

  const int M = kB * kS;  // 4096
  dim3 blk(256);

  // Projections (bf16 outputs for WMMA consumption)
  gemm_bf16_wmma<true><<<dim3(kDM / 64, M / 128), blk, 0, stream>>>(
      query, Wq, bq, Qb, M, kDM, kDM);
  gemm_bf16_wmma<true><<<dim3(kDM / 64, M / 128), blk, 0, stream>>>(
      key_, Wk, bk, Kb, M, kDM, kDM);
  gemm_bf16_wmma<true><<<dim3(kDK / 64, M / 128), blk, 0, stream>>>(
      value, Wv, bv, Vb, M, kDK, kDM);

  // Fused scores -> softmax -> head-average -> avg@V
  attn_fused<<<dim3(kB * (kS / 16)), blk, 0, stream>>>(Qb, Kb, Vb, avg,
                                                       attn_out);

  // Output projection (fp32 out, + bias)
  gemm_bf16_wmma<false><<<dim3(kDM / 64, M / 128), blk, 0, stream>>>(
      attn_out, Wo, bo, out, M, kDM, kDK);
}
